// ImprovedSeq2SeqModel_30571577213742
// MI455X (gfx1250) — matrix operand order repaired and checked by reading, not compiled, not run on hardware
//
#include <hip/hip_runtime.h>
#include <hip/hip_bf16.h>

// ---------------- problem constants ----------------
#define Bq 32
#define Sq 128
#define Tq 64
#define Vq 32000
#define Eq 256
#define Hq 512

typedef __attribute__((ext_vector_type(16))) __bf16 bf16x16;
typedef __attribute__((ext_vector_type(8)))  float  floatx8;
typedef __attribute__((ext_vector_type(4)))  int    i32x4;

union Frag { bf16x16 v; uint4 q[2]; };
union U8   { unsigned short s[8]; uint4 q; };

__device__ __forceinline__ unsigned short f2bf(float f) {
  unsigned u = __float_as_uint(f);
  u += 0x7FFFu + ((u >> 16) & 1u);          // round-to-nearest-even
  return (unsigned short)(u >> 16);
}
__device__ __forceinline__ float sigm(float x) { return 1.0f / (1.0f + expf(-x)); }

__device__ __forceinline__ floatx8 wmma_bf16(const Frag& a, const Frag& b, floatx8 c) {
  return __builtin_amdgcn_wmma_f32_16x16x32_bf16(false, a.v, false, b.v, (short)0, c, false, false);
}

// ---- CDNA5 async direct-to-LDS staging (ASYNCcnt path), with sync fallback ----
#if __has_builtin(__builtin_amdgcn_global_load_async_to_lds_b128) && \
    __has_builtin(__builtin_amdgcn_s_wait_asynccnt)
#define USE_ASYNC_LDS 1
#else
#define USE_ASYNC_LDS 0
#endif

__device__ __forceinline__ void lds_copy16(unsigned short* dst, const unsigned short* src) {
#if USE_ASYNC_LDS
  // builtin signature (probed): (global int4* src, local int4* dst, imm offset, imm cpol)
  __builtin_amdgcn_global_load_async_to_lds_b128(
      (__attribute__((address_space(1))) i32x4*)(src),
      (__attribute__((address_space(3))) i32x4*)(dst), 0, 0);
#else
  *reinterpret_cast<uint4*>(dst) = *reinterpret_cast<const uint4*>(src);
#endif
}
__device__ __forceinline__ void async_fence() {
#if USE_ASYNC_LDS
  __builtin_amdgcn_s_wait_asynccnt(0);
#endif
}

// A-matrix 16x32 bf16 fragment (ISA 7.12.2): lanes 0-15 -> rows, K = {0..7,16..23}; lanes 16-31: +8
__device__ __forceinline__ Frag load_fragA(const unsigned short* base16, int lane) {
  const unsigned short* rp = base16 + (lane & 15) * 32 + ((lane & 16) ? 8 : 0);
  Frag f;
  f.q[0] = *reinterpret_cast<const uint4*>(rp);
  f.q[1] = *reinterpret_cast<const uint4*>(rp + 16);
  return f;
}
// B-matrix 32x16 bf16 fragment: lanes 0-15 -> cols with K = {0..7,16..23}; lanes 16-31: +8
__device__ __forceinline__ Frag load_fragB(const unsigned short* base16, int lane) {
  const unsigned short* rp = base16 + (lane & 15) * 32 + ((lane & 16) ? 8 : 0);
  Frag f;
  f.q[0] = *reinterpret_cast<const uint4*>(rp);
  f.q[1] = *reinterpret_cast<const uint4*>(rp + 16);
  return f;
}

// ================= 64x64-tile bf16 GEMM (skinny M: per-step decoder GEMMs, bridge) ======
// C[m,n] = sum_k A[m,k]*W[n,k] + bias[n]       (out = A @ W.T + b)
__global__ __launch_bounds__(256)
void k_gemm_bf16(const unsigned short* __restrict__ A, int K,
                 const float* __restrict__ W, int ldw,
                 const float* __restrict__ bias,
                 float* __restrict__ C, long ldc,
                 int M, int N, int outBmod, long outStrideB)
{
  __shared__ alignas(16) unsigned short At[64][32];
  __shared__ alignas(16) unsigned short Bt[64][32];
  const int tid = threadIdx.x, lane = tid & 31, wave = tid >> 5;
  const int bM = blockIdx.y * 64, bN = blockIdx.x * 64;
  const int tm = wave >> 1, tn0 = (wave & 1) * 2;
  floatx8 acc0 = {}; floatx8 acc1 = {};
  const int rowT = tid >> 2, colb = (tid & 3) * 8;

  for (int k0 = 0; k0 < K; k0 += 32) {
    { // stage A tile (bf16) via async direct-to-LDS
      int gm = bM + rowT;
      if (gm < M) {
        lds_copy16(&At[rowT][colb], A + (long)gm * K + k0 + colb);
      } else {
        *reinterpret_cast<uint4*>(&At[rowT][colb]) = make_uint4(0u, 0u, 0u, 0u);
      }
    }
    { // stage W tile, fp32 -> bf16 in-flight
      int gn = bN + rowT;
      U8 u;
      if (gn < N) {
        const float* wp = W + (long)gn * ldw + k0 + colb;
        float4 f0 = *reinterpret_cast<const float4*>(wp);
        float4 f1 = *reinterpret_cast<const float4*>(wp + 4);
        if (k0 + 32 < K) __builtin_prefetch(wp + 32, 0, 3);   // global_prefetch_b8
        u.s[0]=f2bf(f0.x); u.s[1]=f2bf(f0.y); u.s[2]=f2bf(f0.z); u.s[3]=f2bf(f0.w);
        u.s[4]=f2bf(f1.x); u.s[5]=f2bf(f1.y); u.s[6]=f2bf(f1.z); u.s[7]=f2bf(f1.w);
      } else { u.q = make_uint4(0u,0u,0u,0u); }
      *reinterpret_cast<uint4*>(&Bt[rowT][colb]) = u.q;
    }
    async_fence();
    __syncthreads();
    Frag fa  = load_fragA(&At[tm * 16][0], lane);
    Frag fb0 = load_fragB(&Bt[tn0 * 16][0], lane);
    Frag fb1 = load_fragB(&Bt[(tn0 + 1) * 16][0], lane);
    acc0 = wmma_bf16(fa, fb0, acc0);
    acc1 = wmma_bf16(fa, fb1, acc1);
    __syncthreads();
  }
  const int mBase = tm * 16 + ((lane & 16) ? 8 : 0);
  const int nl = lane & 15;
  for (int half = 0; half < 2; ++half) {
    floatx8 acc = half ? acc1 : acc0;
    int gn = bN + (tn0 + half) * 16 + nl;
    if (gn >= N) continue;
    float bv = bias ? bias[gn] : 0.0f;
    for (int r = 0; r < 8; ++r) {
      int gm = bM + mBase + r;
      if (gm >= M) continue;
      long idx = (outBmod > 0)
                   ? (long)(gm % outBmod) * outStrideB + (long)(gm / outBmod) * ldc + gn
                   : (long)gm * ldc + gn;
      C[idx] = acc[r] + bv;
    }
  }
}

// ================= 128x128-tile bf16 GEMM (big GEMMs: gi_all, Ue, fc) =================
// 8 waves, each owns a 32x64 sub-tile => 8 WMMAs per K-chunk per wave.
__global__ __launch_bounds__(256)
void k_gemm_bf16_128(const unsigned short* __restrict__ A, int K,
                     const float* __restrict__ W, int ldw,
                     const float* __restrict__ bias,
                     float* __restrict__ C, long ldc,
                     int M, int N, int outBmod, long outStrideB)
{
  __shared__ alignas(16) unsigned short At[128][32];
  __shared__ alignas(16) unsigned short Bt[128][32];
  const int tid = threadIdx.x, lane = tid & 31, wave = tid >> 5;
  const int bM = blockIdx.y * 128, bN = blockIdx.x * 128;
  const int mg = wave >> 1, ng = wave & 1;   // m-tiles 2mg..2mg+1, n-tiles 4ng..4ng+3
  floatx8 acc[2][4] = {};

  for (int k0 = 0; k0 < K; k0 += 32) {
#pragma unroll
    for (int i = 0; i < 2; ++i) {            // 512 16B-chunks per operand, 2 per thread
      int idx = tid * 2 + i;
      int row = idx >> 2, colb = (idx & 3) * 8;
      // A tile: async direct-to-LDS
      int gm = bM + row;
      if (gm < M) {
        lds_copy16(&At[row][colb], A + (long)gm * K + k0 + colb);
      } else {
        *reinterpret_cast<uint4*>(&At[row][colb]) = make_uint4(0u, 0u, 0u, 0u);
      }
      // W tile: fp32 -> bf16
      int gn = bN + row;
      U8 u;
      if (gn < N) {
        const float* wp = W + (long)gn * ldw + k0 + colb;
        float4 f0 = *reinterpret_cast<const float4*>(wp);
        float4 f1 = *reinterpret_cast<const float4*>(wp + 4);
        if (k0 + 32 < K) __builtin_prefetch(wp + 32, 0, 3);
        u.s[0]=f2bf(f0.x); u.s[1]=f2bf(f0.y); u.s[2]=f2bf(f0.z); u.s[3]=f2bf(f0.w);
        u.s[4]=f2bf(f1.x); u.s[5]=f2bf(f1.y); u.s[6]=f2bf(f1.z); u.s[7]=f2bf(f1.w);
      } else { u.q = make_uint4(0u,0u,0u,0u); }
      *reinterpret_cast<uint4*>(&Bt[row][colb]) = u.q;
    }
    async_fence();
    __syncthreads();
    Frag fa[2], fb[4];
#pragma unroll
    for (int mi = 0; mi < 2; ++mi) fa[mi] = load_fragA(&At[(2 * mg + mi) * 16][0], lane);
#pragma unroll
    for (int ni = 0; ni < 4; ++ni) fb[ni] = load_fragB(&Bt[(4 * ng + ni) * 16][0], lane);
#pragma unroll
    for (int mi = 0; mi < 2; ++mi)
#pragma unroll
      for (int ni = 0; ni < 4; ++ni)
        acc[mi][ni] = wmma_bf16(fa[mi], fb[ni], acc[mi][ni]);
    __syncthreads();
  }
  const int nl = lane & 15;
  const int mOff = (lane & 16) ? 8 : 0;
#pragma unroll
  for (int mi = 0; mi < 2; ++mi)
#pragma unroll
    for (int ni = 0; ni < 4; ++ni) {
      int gn = bN + (4 * ng + ni) * 16 + nl;
      if (gn >= N) continue;
      float bv = bias ? bias[gn] : 0.0f;
      for (int r = 0; r < 8; ++r) {
        int gm = bM + (2 * mg + mi) * 16 + mOff + r;
        if (gm >= M) continue;
        long idx = (outBmod > 0)
                     ? (long)(gm % outBmod) * outStrideB + (long)(gm / outBmod) * ldc + gn
                     : (long)gm * ldc + gn;
        C[idx] = acc[mi][ni][r] + bv;
      }
    }
}

// ================= fused GRU recurrent step =================
// gh = h @ Whh.T (3 gate blocks via WMMA), then full GRU elementwise.
__global__ __launch_bounds__(256)
void k_gru_step(const unsigned short* __restrict__ hIn,   // [ndir][B][H] bf16
                unsigned short* __restrict__ hOut,        // [ndir][B][H] bf16
                float* __restrict__ hF,                   // [ndir][B][H] fp32, in/out
                const float* __restrict__ Whh, long whhDirStride,   // [3H][H]
                const float* __restrict__ bhh, int bhhDirStride,    // [3H]
                const float* __restrict__ gi,  long giDirStride,    // [S*B][3H]
                int t, int Slen,
                float* __restrict__ y,                    // nullable: [S][B][2H]
                unsigned short* __restrict__ cat, int catRow0, int catLd)
{
  __shared__ alignas(16) unsigned short Ht[32][32];
  __shared__ alignas(16) unsigned short Wt[3][64][32];
  const int tid = threadIdx.x, lane = tid & 31, wave = tid >> 5;
  const int dir = blockIdx.y;
  const int ts = dir ? (Slen - 1 - t) : t;
  const int bN = blockIdx.x * 64;                  // hidden-unit slice
  const unsigned short* hInD = hIn + (long)dir * Bq * Hq;
  unsigned short* hOutD = hOut + (long)dir * Bq * Hq;
  float* hFD = hF + (long)dir * Bq * Hq;
  const float* WhhD = Whh + (long)dir * whhDirStride;
  const float* bhhD = bhh + (long)dir * bhhDirStride;
  const float* giRow = gi + (long)dir * giDirStride + (long)ts * Bq * 3 * Hq;

  const int tm = wave >> 2, tn = wave & 3;         // 2 m-tiles x 4 n-tiles, 3 gates each
  floatx8 aR = {}; floatx8 aZ = {}; floatx8 aN = {};
  const int rowT = tid >> 2, colb = (tid & 3) * 8;

  for (int k0 = 0; k0 < Hq; k0 += 32) {
    if (tid < 128) {                               // h tile: 32 x 32 bf16, async to LDS
      lds_copy16(&Ht[rowT][colb], hInD + (long)rowT * Hq + k0 + colb);
    }
    for (int g = 0; g < 3; ++g) {                  // Whh tiles (fp32 -> bf16)
      const float* wp = WhhD + (long)(g * Hq + bN + rowT) * Hq + k0 + colb;
      float4 f0 = *reinterpret_cast<const float4*>(wp);
      float4 f1 = *reinterpret_cast<const float4*>(wp + 4);
      U8 u;
      u.s[0]=f2bf(f0.x); u.s[1]=f2bf(f0.y); u.s[2]=f2bf(f0.z); u.s[3]=f2bf(f0.w);
      u.s[4]=f2bf(f1.x); u.s[5]=f2bf(f1.y); u.s[6]=f2bf(f1.z); u.s[7]=f2bf(f1.w);
      *reinterpret_cast<uint4*>(&Wt[g][rowT][colb]) = u.q;
    }
    async_fence();
    __syncthreads();
    Frag fa = load_fragA(&Ht[tm * 16][0], lane);
    Frag fr = load_fragB(&Wt[0][tn * 16][0], lane);
    Frag fz = load_fragB(&Wt[1][tn * 16][0], lane);
    Frag fn = load_fragB(&Wt[2][tn * 16][0], lane);
    aR = wmma_bf16(fa, fr, aR);
    aZ = wmma_bf16(fa, fz, aZ);
    aN = wmma_bf16(fa, fn, aN);
    __syncthreads();
  }

  const int j = bN + tn * 16 + (lane & 15);        // hidden unit
  const int mBase = tm * 16 + ((lane & 16) ? 8 : 0);
  const float brv = bhhD[j], bzv = bhhD[Hq + j], bnv = bhhD[2 * Hq + j];
  for (int r = 0; r < 8; ++r) {
    int m = mBase + r;                             // batch row (0..31 exactly)
    const float* gim = giRow + (long)m * 3 * Hq;
    float ghr = aR[r] + brv, ghz = aZ[r] + bzv, ghn = aN[r] + bnv;
    float rg = sigm(gim[j] + ghr);
    float zg = sigm(gim[Hq + j] + ghz);
    float ng = tanhf(gim[2 * Hq + j] + rg * ghn);
    float hp = hFD[(long)m * Hq + j];
    float hn2 = (1.0f - zg) * ng + zg * hp;
    hFD[(long)m * Hq + j] = hn2;
    hOutD[(long)m * Hq + j] = f2bf(hn2);
    if (y)   y[((long)ts * Bq + m) * (2 * Hq) + dir * Hq + j] = hn2;
    if (cat) cat[(long)(catRow0 + m) * catLd + j] = f2bf(hn2);
  }
}

// ================= attention (per-batch-row block) =================
__global__ __launch_bounds__(256)
void k_attn(const float* __restrict__ q,          // [B,H], includes attn_b
            const float* __restrict__ Ue,         // [S*B, H] precomputed enc part
            const float* __restrict__ av,         // [H]
            const float* __restrict__ enc,        // [S*B, 2H] encoder outputs
            const unsigned short* __restrict__ tgtE, // [T*B, E] bf16
            unsigned short* __restrict__ rin,     // [B, E+2H] bf16
            unsigned short* __restrict__ cat,     // [T*B, 3H] bf16
            int t)
{
  __shared__ float sc[Sq];
  __shared__ float invsum;
  const int b = blockIdx.x;
  const int tid = threadIdx.x, lane = tid & 31, wave = tid >> 5;
  const float* qq = q + (long)b * Hq;
  for (int s = wave; s < Sq; s += 8) {
    const float* ue = Ue + ((long)s * Bq + b) * Hq;
    float p = 0.0f;
    for (int h = lane; h < Hq; h += 32)
      p += tanhf(qq[h] + ue[h]) * av[h];
    for (int off = 16; off; off >>= 1) p += __shfl_xor(p, off, 32);
    if (lane == 0) sc[s] = p;
  }
  __syncthreads();
  if (tid == 0) {
    float mx = sc[0];
    for (int s = 1; s < Sq; ++s) mx = fmaxf(mx, sc[s]);
    float sm = 0.0f;
    for (int s = 0; s < Sq; ++s) { float e = expf(sc[s] - mx); sc[s] = e; sm += e; }
    invsum = 1.0f / sm;
  }
  __syncthreads();
  const float inv = invsum;
  for (int d = tid; d < 2 * Hq; d += 256) {
    float a = 0.0f;
    for (int s = 0; s < Sq; ++s)
      a += sc[s] * enc[((long)s * Bq + b) * (2 * Hq) + d];
    unsigned short bv = f2bf(a * inv);
    rin[(long)b * (Eq + 2 * Hq) + Eq + d] = bv;
    cat[((long)t * Bq + b) * (3 * Hq) + Hq + d] = bv;
  }
  if (tid < Eq)
    rin[(long)b * (Eq + 2 * Hq) + tid] = tgtE[((long)t * Bq + b) * Eq + tid];
}

// ================= small helpers =================
__global__ void k_embed(const int* __restrict__ tok, const float* __restrict__ emb,
                        unsigned short* __restrict__ out, int L) {
  int idx = blockIdx.x;
  int b = idx % Bq, l = idx / Bq;
  int tk = tok[b * L + l];
  out[((long)l * Bq + b) * Eq + threadIdx.x] = f2bf(emb[(long)tk * Eq + threadIdx.x]);
}

__global__ void k_f2b(const float* __restrict__ in, unsigned short* __restrict__ out, long n) {
  long i = (long)blockIdx.x * blockDim.x + threadIdx.x;
  if (i < n) out[i] = f2bf(in[i]);
}

__global__ void k_zero(unsigned int* __restrict__ p, long n) {
  long i = (long)blockIdx.x * blockDim.x + threadIdx.x;
  if (i < n) p[i] = 0u;
}

__global__ void k_finals(const float* __restrict__ y, unsigned short* __restrict__ fin, int L) {
  int b = blockIdx.x;
  for (int c = threadIdx.x; c < 2 * Hq; c += 256) {
    // forward final state = y[S-1, b, 0:H]; backward final = y[0, b, H:2H]
    float v = (c < Hq) ? y[((long)(Sq - 1) * Bq + b) * (2 * Hq) + c]
                       : y[((long)b) * (2 * Hq) + c];
    fin[((long)L * Bq + b) * (2 * Hq) + c] = f2bf(v);
  }
}

// ================= host orchestration =================
extern "C" void kernel_launch(void* const* d_in, const int* in_sizes, int n_in,
                              void* d_out, int out_size, void* d_ws, size_t ws_size,
                              hipStream_t stream) {
  const int*   src      = (const int*)  d_in[0];
  const int*   tgt      = (const int*)  d_in[1];
  const float* emb      = (const float*)d_in[2];
  const float* eWih0    = (const float*)d_in[3];
  const float* eWhh0    = (const float*)d_in[4];
  const float* ebih0    = (const float*)d_in[5];
  const float* ebhh0    = (const float*)d_in[6];
  const float* eWih     = (const float*)d_in[7];
  const float* eWhh     = (const float*)d_in[8];
  const float* ebih     = (const float*)d_in[9];
  const float* ebhh     = (const float*)d_in[10];
  const float* bridgeW  = (const float*)d_in[11];
  const float* bridgeB  = (const float*)d_in[12];
  const float* attnW    = (const float*)d_in[13];
  const float* attnB    = (const float*)d_in[14];
  const float* attnV    = (const float*)d_in[15];
  const float* dWih0    = (const float*)d_in[16];
  const float* dWhh0    = (const float*)d_in[17];
  const float* dbih0    = (const float*)d_in[18];
  const float* dbhh0    = (const float*)d_in[19];
  const float* dWih     = (const float*)d_in[20];
  const float* dWhh     = (const float*)d_in[21];
  const float* dbih     = (const float*)d_in[22];
  const float* dbhh     = (const float*)d_in[23];
  const float* fcW      = (const float*)d_in[24];
  const float* fcB      = (const float*)d_in[25];
  float* out = (float*)d_out;

  // ---- workspace carve-out (~93 MB) ----
  char* w = (char*)d_ws;
  auto carve = [&](size_t bytes) { void* p = (void*)w; w += (bytes + 255) & ~(size_t)255; return p; };
  unsigned short* xsB   = (unsigned short*)carve((size_t)Sq*Bq*2*Hq*2);     // layer input, bf16
  float*          yF    = (float*)         carve((size_t)Sq*Bq*2*Hq*4);     // layer output / enc_out
  float*          giAll = (float*)         carve((size_t)2*Sq*Bq*3*Hq*4);   // hoisted input gates
  float*          Ue    = (float*)         carve((size_t)Sq*Bq*Hq*4);       // attn enc projection
  unsigned short* tgtE  = (unsigned short*)carve((size_t)Tq*Bq*Eq*2);
  unsigned short* catB  = (unsigned short*)carve((size_t)Tq*Bq*3*Hq*2);     // [h2n,ctx] for fc
  unsigned short* finB  = (unsigned short*)carve((size_t)3*Bq*2*Hq*2);
  float*          hid   = (float*)         carve((size_t)3*Bq*Hq*4);        // bridge out = dec h fp32
  unsigned short* hdecB = (unsigned short*)carve((size_t)2*3*Bq*Hq*2);      // dec h bf16 ping-pong
  float*          hencF = (float*)         carve((size_t)2*Bq*Hq*4);        // enc h fp32 (2 dirs)
  unsigned short* hencB = (unsigned short*)carve((size_t)2*2*Bq*Hq*2);      // enc h bf16 ping-pong
  float*          qBuf  = (float*)         carve((size_t)Bq*Hq*4);
  unsigned short* rinB  = (unsigned short*)carve((size_t)Bq*(Eq+2*Hq)*2);
  float*          giD   = (float*)         carve((size_t)Bq*3*Hq*4);

  auto gemm = [&](const unsigned short* A, int K, const float* W, int ldw,
                  const float* bias, float* C, long ldc, int M, int N,
                  int outBmod = 0, long outStrideB = 0) {
    dim3 g((N + 63) / 64, (M + 63) / 64);
    k_gemm_bf16<<<g, 256, 0, stream>>>(A, K, W, ldw, bias, C, ldc, M, N, outBmod, outStrideB);
  };
  auto gemm128 = [&](const unsigned short* A, int K, const float* W, int ldw,
                     const float* bias, float* C, long ldc, int M, int N,
                     int outBmod = 0, long outStrideB = 0) {
    dim3 g((N + 127) / 128, (M + 127) / 128);
    k_gemm_bf16_128<<<g, 256, 0, stream>>>(A, K, W, ldw, bias, C, ldc, M, N, outBmod, outStrideB);
  };

  // ================= ENCODER =================
  for (int L = 0; L < 3; ++L) {
    int D = (L == 0) ? Eq : 2 * Hq;
    if (L == 0) k_embed<<<Bq * Sq, 256, 0, stream>>>(src, emb, xsB, Sq);
    else { long n = (long)Sq*Bq*2*Hq; k_f2b<<<(int)((n+255)/256), 256, 0, stream>>>(yF, xsB, n); }

    const float* WihL = (L == 0) ? eWih0 : eWih + (size_t)(L-1)*2*3*Hq*2*Hq;
    const float* bihL = (L == 0) ? ebih0 : ebih + (size_t)(L-1)*2*3*Hq;
    const float* WhhL = (L == 0) ? eWhh0 : eWhh + (size_t)(L-1)*2*3*Hq*Hq;
    const float* bhhL = (L == 0) ? ebhh0 : ebhh + (size_t)(L-1)*2*3*Hq;

    // hoisted input GEMM: gi_all[dir] = xs @ Wih.T + bih   ([S*B, 3H])
    for (int dir = 0; dir < 2; ++dir)
      gemm128(xsB, D, WihL + (size_t)dir*3*Hq*D, D, bihL + (size_t)dir*3*Hq,
              giAll + (size_t)dir*Sq*Bq*3*Hq, 3*Hq, Sq*Bq, 3*Hq);

    { long n = (2L*Bq*Hq*4)/4;            k_zero<<<(int)((n+255)/256),256,0,stream>>>((unsigned int*)hencF, n); }
    { long n = (2L*Bq*Hq*2)/4;            k_zero<<<(int)((n+255)/256),256,0,stream>>>((unsigned int*)hencB, n); }

    for (int t = 0; t < Sq; ++t) {
      int in = t & 1;
      k_gru_step<<<dim3(Hq/64, 2), 256, 0, stream>>>(
          hencB + (size_t)in*2*Bq*Hq, hencB + (size_t)(1-in)*2*Bq*Hq, hencF,
          WhhL, (long)3*Hq*Hq, bhhL, 3*Hq,
          giAll, (long)Sq*Bq*3*Hq, t, Sq,
          yF, nullptr, 0, 0);
    }
    k_finals<<<Bq, 256, 0, stream>>>(yF, finB, L);
  }

  // bridge: hidden[3,B,H] = finals @ bridge_W.T + b
  gemm(finB, 2*Hq, bridgeW, 2*Hq, bridgeB, hid, Hq, 3*Bq, Hq);

  // enc_out bf16; Ue = enc_out @ attn_W[:, H:].T  (step-invariant attention half)
  { long n = (long)Sq*Bq*2*Hq; k_f2b<<<(int)((n+255)/256), 256, 0, stream>>>(yF, xsB, n); }
  gemm128(xsB, 2*Hq, attnW + Hq, 3*Hq, nullptr, Ue, Hq, Sq*Bq, Hq);

  k_embed<<<Bq * Tq, 256, 0, stream>>>(tgt, emb, tgtE, Tq);
  { long n = (long)3*Bq*Hq; k_f2b<<<(int)((n+255)/256), 256, 0, stream>>>(hid, hdecB, n); }

  // ================= DECODER (recurrent, fc deferred) =================
  for (int t = 0; t < Tq; ++t) {
    int in = t & 1, outp = 1 - in;
    unsigned short* hBin  = hdecB + (size_t)in  *3*Bq*Hq;
    unsigned short* hBout = hdecB + (size_t)outp*3*Bq*Hq;

    // q = h2 @ attn_W[:, :H].T + attn_b
    gemm(hBin + (size_t)2*Bq*Hq, Hq, attnW, 3*Hq, attnB, qBuf, Hq, Bq, Hq);
    k_attn<<<Bq, 256, 0, stream>>>(qBuf, Ue, attnV, yF, tgtE, rinB, catB, t);

    // cell 0: rin = [x_t, ctx]  (K = E + 2H = 1280)
    gemm(rinB, Eq + 2*Hq, dWih0, Eq + 2*Hq, dbih0, giD, 3*Hq, Bq, 3*Hq);
    k_gru_step<<<dim3(Hq/64, 1), 256, 0, stream>>>(
        hBin, hBout, hid, dWhh0, 0, dbhh0, 0, giD, 0, 0, 1, nullptr, nullptr, 0, 0);

    // cell 1: input = h0n (just written to hBout layer0)
    gemm(hBout, Hq, dWih, Hq, dbih, giD, 3*Hq, Bq, 3*Hq);
    k_gru_step<<<dim3(Hq/64, 1), 256, 0, stream>>>(
        hBin + (size_t)Bq*Hq, hBout + (size_t)Bq*Hq, hid + (size_t)Bq*Hq,
        dWhh, 0, dbhh, 0, giD, 0, 0, 1, nullptr, nullptr, 0, 0);

    // cell 2: input = h1n; writes h2n into cat rows [t*B .. t*B+31], cols 0..H
    gemm(hBout + (size_t)Bq*Hq, Hq, dWih + (size_t)3*Hq*Hq, Hq, dbih + 3*Hq, giD, 3*Hq, Bq, 3*Hq);
    k_gru_step<<<dim3(Hq/64, 1), 256, 0, stream>>>(
        hBin + (size_t)2*Bq*Hq, hBout + (size_t)2*Bq*Hq, hid + (size_t)2*Bq*Hq,
        dWhh + (size_t)3*Hq*Hq, 0, dbhh + 3*Hq, 0, giD, 0, 0, 1,
        nullptr, catB, t * Bq, 3*Hq);
  }

  // ===== batched output projection: [T*B, 3H] @ fc_W.T + fc_b -> out[b, t, v] =====
  gemm128(catB, 3*Hq, fcW, 3*Hq, fcB, out, Vq, Tq*Bq, Vq, /*outBmod=*/Bq, /*strideB=*/(long)Tq*Vq);
}
